// TransformerWithMoE_2276332667121
// MI455X (gfx1250) — matrix operand order repaired and checked by reading, not compiled, not run on hardware
//
#include <hip/hip_runtime.h>
#include <cstdint>

// ---------------------------------------------------------------------------
// Model dims
// ---------------------------------------------------------------------------
#define B_  8
#define S_  2048
#define IN_ 128
#define D_  512
#define FF_ 2048
#define L_  3
#define H_  8
#define DH_ 64
#define NEXP_ 8
#define M_  (B_ * S_)          // 16384 tokens

typedef __attribute__((ext_vector_type(8)))  float  v8f;
typedef __attribute__((ext_vector_type(8)))  __bf16 v8bf;
typedef __attribute__((ext_vector_type(16))) __bf16 v16bf;

union F16B { v16bf v; __bf16 e[16]; };
union F8F  { v8f   v; float  e[8];  };

static __device__ __forceinline__ v8f vzero8() {
  v8f z = {0.f, 0.f, 0.f, 0.f, 0.f, 0.f, 0.f, 0.f};
  return z;
}
static __device__ __forceinline__ v16bf cat8(v8bf lo, v8bf hi) {
  return __builtin_shufflevector(lo, hi, 0,1,2,3,4,5,6,7,8,9,10,11,12,13,14,15);
}

// ---------------------------------------------------------------------------
// f32 -> bf16 convert
// ---------------------------------------------------------------------------
__global__ void k_cvt_bf16(const float* __restrict__ src, __bf16* __restrict__ dst, size_t n) {
  size_t i = (size_t)blockIdx.x * 256 + threadIdx.x;
  if (i < n) dst[i] = (__bf16)src[i];
}

// ---------------------------------------------------------------------------
// Generic WMMA GEMM:  C[M,N] = epi( A[M,K](bf16) x W[N,K]^T(bf16) + bias )
//  - 8 waves / block, wave tile 32x64, block tile 128x128
//  - act: 0 none, 1 exact GELU, 2 ReLU
//  - resid  != null : += resid[m*N+n]
//  - rowscale != null : Cf[m*N+n] += rowscale[m*rs_stride] * val  (MoE accumulate)
// ---------------------------------------------------------------------------
__global__ __launch_bounds__(256)
void k_gemm_bf16(const __bf16* __restrict__ A, const __bf16* __restrict__ W,
                 const float* __restrict__ bias,
                 float* __restrict__ Cf, __bf16* __restrict__ Cb,
                 const float* __restrict__ resid,
                 const float* __restrict__ rowscale, int rs_stride,
                 int M, int N, int K, int act)
{
  const int lane = threadIdx.x & 31;
  const int wave = threadIdx.x >> 5;
  const int wm = wave & 3;          // 4 waves along M
  const int wn = wave >> 2;         // 2 waves along N
  const int m0 = blockIdx.y * 128 + wm * 32;
  const int n0 = blockIdx.x * 128 + wn * 64;
  const int half = lane >> 4;       // 0 | 1
  const int l16  = lane & 15;

  v8f acc[2][4];
  #pragma unroll
  for (int i = 0; i < 2; ++i)
    #pragma unroll
    for (int j = 0; j < 4; ++j) acc[i][j] = vzero8();

  const __bf16* Arow[2];
  Arow[0] = A + (size_t)(m0 + l16) * K;
  Arow[1] = A + (size_t)(m0 + 16 + l16) * K;
  const __bf16* Wrow[4];
  #pragma unroll
  for (int j = 0; j < 4; ++j) {
    int n = n0 + j * 16 + l16;
    if (n >= N) n = N - 1;                 // clamp: loads stay in-bounds, stores guarded
    Wrow[j] = W + (size_t)n * K;
  }

  for (int k = 0; k < K; k += 32) {
    v16bf a[2], b[4];
    #pragma unroll
    for (int i = 0; i < 2; ++i) {
      const __bf16* p = Arow[i] + k + half * 8;       // K chunks {+0..7, +16..23}
      a[i] = cat8(*(const v8bf*)p, *(const v8bf*)(p + 16));
    }
    #pragma unroll
    for (int j = 0; j < 4; ++j) {
      const __bf16* q = Wrow[j] + k + half * 8;        // K chunks {+0..7, +16..23}
      b[j] = cat8(*(const v8bf*)q, *(const v8bf*)(q + 16));
    }
    #pragma unroll
    for (int i = 0; i < 2; ++i)
      #pragma unroll
      for (int j = 0; j < 4; ++j)
        acc[i][j] = __builtin_amdgcn_wmma_f32_16x16x32_bf16(
            false, a[i], false, b[j], (short)0, acc[i][j], false, false);
  }

  #pragma unroll
  for (int i = 0; i < 2; ++i) {
    #pragma unroll
    for (int j = 0; j < 4; ++j) {
      int n = n0 + j * 16 + l16;
      if (n >= N) continue;
      float bs = bias ? bias[n] : 0.f;
      F8F f; f.v = acc[i][j];
      #pragma unroll
      for (int v = 0; v < 8; ++v) {
        int m = m0 + i * 16 + half * 8 + v;           // C layout: row = v + 8*half
        float val = f.e[v] + bs;
        if (act == 1)      val = 0.5f * val * (1.f + erff(val * 0.70710678118f));
        else if (act == 2) val = fmaxf(val, 0.f);
        size_t idx = (size_t)m * N + n;
        if (resid) val += resid[idx];
        if (rowscale) {
          Cf[idx] += rowscale[(size_t)m * rs_stride] * val;
        } else {
          if (Cf) Cf[idx] = val;
          if (Cb) Cb[idx] = (__bf16)val;
        }
      }
    }
  }
}

// ---------------------------------------------------------------------------
// Positional encoding add (in place on f32 h [M, D])
// ---------------------------------------------------------------------------
__global__ void k_add_pe(float* __restrict__ h) {
  size_t i = (size_t)blockIdx.x * 256 + threadIdx.x;
  if (i >= (size_t)M_ * D_) return;
  int d = (int)(i % D_);
  size_t t = i / D_;
  int s = (int)(t % S_);
  int j = d >> 1;
  float div = __expf((float)(2 * j) * (-9.210340371976f / (float)D_)); // -ln(10000)/D
  float a = (float)s * div;
  h[i] += (d & 1) ? __cosf(a) : __sinf(a);
}

// ---------------------------------------------------------------------------
// Flash attention: one wave per (b, h, 16-query tile). qkv bf16 [B*S, 3D]
// (bias fused by the qkv GEMM). Writes bf16 attention output [B*S, D].
// V tiles are double-buffered into LDS with GLOBAL_LOAD_ASYNC_TO_LDS_B128
// (ASYNCcnt-tracked, overlapped with QK^T + softmax), then gathered from LDS.
// ---------------------------------------------------------------------------
#define VSTRIDE 68   // padded LDS row stride (elems): disjoint bank groups per half-wave

__global__ __launch_bounds__(32)
void k_flash_attn(const __bf16* __restrict__ qkv, __bf16* __restrict__ out)
{
  __shared__ __bf16 Plds[16][36];              // 16x32 P tile (bf16), padded
  __shared__ __bf16 Vlds[2][32 * VSTRIDE];     // double-buffered 32x64 V tiles

  const int lane = threadIdx.x & 31;
  const int l16  = lane & 15;
  const int half = lane >> 4;
  const int q0 = blockIdx.x * 16;
  const int h  = blockIdx.y;
  const int b  = blockIdx.z;
  const size_t rs = 3 * D_;
  const __bf16* base = qkv + (size_t)b * S_ * rs;

  // Q A-fragments (kept resident): 16 x 64, two K=32 fragments
  F16B aq[2];
  {
    const __bf16* qrow = base + (size_t)(q0 + l16) * rs + h * DH_;
    #pragma unroll
    for (int f = 0; f < 2; ++f) {
      const __bf16* p = qrow + f * 32 + half * 8;
      aq[f].v = cat8(*(const v8bf*)p, *(const v8bf*)(p + 16));
    }
  }

  F8F o[4];
  #pragma unroll
  for (int j = 0; j < 4; ++j) o[j].v = vzero8();
  float mrow[8], lrow[8];
  #pragma unroll
  for (int v = 0; v < 8; ++v) { mrow[v] = -3.0e38f; lrow[v] = 0.f; }

  const float scale = 0.125f;                  // 1/sqrt(64)

  // async V-tile stage: each lane copies its key row (64 bf16) in 8x16B chunks
  auto issueV = [&](int kb, int buf) {
    const __bf16* vrow = base + (size_t)(kb + lane) * rs + 2 * D_ + h * DH_;
    uint64_t g  = (uint64_t)(uintptr_t)vrow;
    uint32_t l0 = (uint32_t)(uintptr_t)&Vlds[buf][lane * VSTRIDE];
    #pragma unroll
    for (int c = 0; c < 8; ++c) {
      asm volatile("global_load_async_to_lds_b128 %0, %1, off"
                   :: "v"(l0 + (uint32_t)(c * 16)), "v"(g + (uint64_t)(c * 16))
                   : "memory");
    }
  };

  issueV(0, 0);

  for (int kb = 0; kb < S_; kb += 32) {
    const int buf = (kb >> 5) & 1;
    const bool more = (kb + 32) < S_;
    if (more) issueV(kb + 32, buf ^ 1);        // prefetch next tile while computing

    // --- scores S = Q K^T (16 x 32) ---
    v16bf bk[2][2];
    #pragma unroll
    for (int nf = 0; nf < 2; ++nf) {
      const __bf16* krow = base + (size_t)(kb + nf * 16 + l16) * rs + D_ + h * DH_;
      #pragma unroll
      for (int f = 0; f < 2; ++f) {
        const __bf16* p = krow + f * 32 + half * 8;
        bk[nf][f] = cat8(*(const v8bf*)p, *(const v8bf*)(p + 16));
      }
    }
    F8F s[2];
    #pragma unroll
    for (int nf = 0; nf < 2; ++nf) {
      v8f t = vzero8();
      t = __builtin_amdgcn_wmma_f32_16x16x32_bf16(false, aq[0].v, false, bk[nf][0], (short)0, t, false, false);
      t = __builtin_amdgcn_wmma_f32_16x16x32_bf16(false, aq[1].v, false, bk[nf][1], (short)0, t, false, false);
      s[nf].v = t;
    }

    // --- online softmax (row stats across the 16-lane half groups) ---
    #pragma unroll
    for (int v = 0; v < 8; ++v) {
      float s0 = s[0].e[v] * scale, s1 = s[1].e[v] * scale;
      float t = fmaxf(s0, s1);
      #pragma unroll
      for (int msk = 1; msk <= 8; msk <<= 1) t = fmaxf(t, __shfl_xor(t, msk, 32));
      float nm = fmaxf(mrow[v], t);
      float corr = __expf(mrow[v] - nm);
      float p0 = __expf(s0 - nm), p1 = __expf(s1 - nm);
      float rsum = p0 + p1;
      #pragma unroll
      for (int msk = 1; msk <= 8; msk <<= 1) rsum += __shfl_xor(rsum, msk, 32);
      lrow[v] = lrow[v] * corr + rsum;
      mrow[v] = nm;
      #pragma unroll
      for (int j = 0; j < 4; ++j) o[j].e[v] *= corr;
      int r = v + 8 * half;
      Plds[r][l16]      = (__bf16)p0;
      Plds[r][16 + l16] = (__bf16)p1;
    }
    __syncthreads();

    // --- re-layout P (C-frag) -> A-frag via LDS ---
    F16B pa;
    #pragma unroll
    for (int j = 0; j < 8; ++j) {
      pa.e[j]     = Plds[l16][half * 8 + j];
      pa.e[8 + j] = Plds[l16][half * 8 + 16 + j];
    }
    __syncthreads();

    // --- wait for current V tile (older 8 async copies; next 8 may stay in flight) ---
    if (more) asm volatile("s_wait_asynccnt 0x8" ::: "memory");
    else      asm volatile("s_wait_asynccnt 0x0" ::: "memory");

    // --- O += P V  (V B-frags gathered from LDS) ---
    #pragma unroll
    for (int nf = 0; nf < 4; ++nf) {
      F16B bv;
      const __bf16* vb = &Vlds[buf][(half * 8) * VSTRIDE + nf * 16 + l16];
      #pragma unroll
      for (int t2 = 0; t2 < 16; ++t2) bv.e[t2] = vb[(t2 + (t2 & 8)) * VSTRIDE];
      o[nf].v = __builtin_amdgcn_wmma_f32_16x16x32_bf16(false, pa.v, false, bv.v, (short)0, o[nf].v, false, false);
    }
  }

  // --- finalize + store ---
  float inv[8];
  #pragma unroll
  for (int v = 0; v < 8; ++v) inv[v] = 1.f / lrow[v];
  #pragma unroll
  for (int nf = 0; nf < 4; ++nf)
    #pragma unroll
    for (int v = 0; v < 8; ++v) {
      int q = q0 + v + 8 * half;
      int c = h * DH_ + nf * 16 + l16;
      out[(size_t)(b * S_ + q) * D_ + c] = (__bf16)(o[nf].e[v] * inv[v]);
    }
}

// ---------------------------------------------------------------------------
// LayerNorm (f32 in) -> bf16 out. One block (256 thr) per row.
// ---------------------------------------------------------------------------
__global__ __launch_bounds__(256)
void k_layernorm_bf16(const float* __restrict__ x, const float* __restrict__ w,
                      const float* __restrict__ bia, __bf16* __restrict__ out)
{
  const int row = blockIdx.x;
  const float* xr = x + (size_t)row * D_;
  float s = 0.f, sq = 0.f;
  for (int i = threadIdx.x; i < D_; i += 256) { float v = xr[i]; s += v; sq += v * v; }
  #pragma unroll
  for (int m = 1; m < 32; m <<= 1) { s += __shfl_xor(s, m, 32); sq += __shfl_xor(sq, m, 32); }
  __shared__ float ws[8], wsq[8];
  int wave = threadIdx.x >> 5, lane = threadIdx.x & 31;
  if (lane == 0) { ws[wave] = s; wsq[wave] = sq; }
  __syncthreads();
  if (threadIdx.x < 32) {
    float a = (lane < 8) ? ws[lane] : 0.f;
    float c = (lane < 8) ? wsq[lane] : 0.f;
    #pragma unroll
    for (int m = 1; m < 8; m <<= 1) { a += __shfl_xor(a, m, 32); c += __shfl_xor(c, m, 32); }
    if (lane == 0) { ws[0] = a; wsq[0] = c; }
  }
  __syncthreads();
  float mean = ws[0] / (float)D_;
  float var  = wsq[0] / (float)D_ - mean * mean;
  float r = rsqrtf(var + 1e-5f);
  for (int i = threadIdx.x; i < D_; i += 256)
    out[(size_t)row * D_ + i] = (__bf16)((xr[i] - mean) * r * w[i] + bia[i]);
}

// ---------------------------------------------------------------------------
// MoE gating: one wave per token. softmax over 8 logits, top-2, L1 renorm.
// ---------------------------------------------------------------------------
__global__ __launch_bounds__(32)
void k_gate(const float* __restrict__ h, const float* __restrict__ gw,
            const float* __restrict__ gb, float* __restrict__ gated)
{
  const int t = blockIdx.x;
  const int lane = threadIdx.x;
  const float* hr = h + (size_t)t * D_;
  float logit[NEXP_];
  #pragma unroll
  for (int n = 0; n < NEXP_; ++n) {
    float s = 0.f;
    for (int d = lane; d < D_; d += 32) s += hr[d] * gw[n * D_ + d];
    #pragma unroll
    for (int m = 1; m < 32; m <<= 1) s += __shfl_xor(s, m, 32);
    logit[n] = s + gb[n];
  }
  float mx = logit[0];
  #pragma unroll
  for (int n = 1; n < NEXP_; ++n) mx = fmaxf(mx, logit[n]);
  float p[NEXP_], sum = 0.f;
  #pragma unroll
  for (int n = 0; n < NEXP_; ++n) { p[n] = __expf(logit[n] - mx); sum += p[n]; }
  #pragma unroll
  for (int n = 0; n < NEXP_; ++n) p[n] /= sum;
  int i1 = 0;
  #pragma unroll
  for (int n = 1; n < NEXP_; ++n) if (p[n] > p[i1]) i1 = n;
  int i2 = (i1 == 0) ? 1 : 0;
  #pragma unroll
  for (int n = 0; n < NEXP_; ++n) { if (n != i1 && p[n] > p[i2]) i2 = n; }
  float norm = fmaxf(p[i1] + p[i2], 1e-12f);
  if (lane == 0) {
    #pragma unroll
    for (int n = 0; n < NEXP_; ++n)
      gated[(size_t)t * NEXP_ + n] = (n == i1 || n == i2) ? p[n] / norm : 0.f;
  }
}

// ---------------------------------------------------------------------------
// Host orchestration
// ---------------------------------------------------------------------------
extern "C" void kernel_launch(void* const* d_in, const int* in_sizes, int n_in,
                              void* d_out, int out_size, void* d_ws, size_t ws_size,
                              hipStream_t stream) {
  (void)in_sizes; (void)n_in; (void)out_size; (void)ws_size;
  const float* x     = (const float*)d_in[0];
  const float* in_w  = (const float*)d_in[1];
  const float* in_b  = (const float*)d_in[2];
  const float* qkv_w = (const float*)d_in[3];
  const float* qkv_b = (const float*)d_in[4];
  const float* out_w = (const float*)d_in[5];
  const float* out_b = (const float*)d_in[6];
  const float* ln1_w = (const float*)d_in[7];
  const float* ln1_b = (const float*)d_in[8];
  const float* ln2_w = (const float*)d_in[9];
  const float* ln2_b = (const float*)d_in[10];
  const float* ff1_w = (const float*)d_in[11];
  const float* ff1_b = (const float*)d_in[12];
  const float* ff2_w = (const float*)d_in[13];
  const float* ff2_b = (const float*)d_in[14];
  const float* gate_w = (const float*)d_in[15];
  const float* gate_b = (const float*)d_in[16];
  const float* e1_w  = (const float*)d_in[17];
  const float* e1_b  = (const float*)d_in[18];
  const float* e2_w  = (const float*)d_in[19];
  const float* e2_b  = (const float*)d_in[20];
  const float* fn_w  = (const float*)d_in[21];
  const float* fn_b  = (const float*)d_in[22];
  const float* hc_w  = (const float*)d_in[23];
  const float* hc_b  = (const float*)d_in[24];
  const float* hb_w  = (const float*)d_in[25];
  const float* hb_b  = (const float*)d_in[26];

  // ---- workspace layout (bytes) ----
  char* ws = (char*)d_ws;
  size_t off = 0;
  auto alloc = [&](size_t bytes) { size_t o = off; off += (bytes + 255) & ~(size_t)255; return o; };
  const size_t n_in_w  = (size_t)D_ * IN_;
  const size_t n_qkv_w = (size_t)L_ * 3 * D_ * D_;
  const size_t n_out_w = (size_t)L_ * D_ * D_;
  const size_t n_ff1_w = (size_t)L_ * FF_ * D_;
  const size_t n_ff2_w = (size_t)L_ * D_ * FF_;
  const size_t n_e1_w  = (size_t)NEXP_ * FF_ * D_;
  const size_t n_e2_w  = (size_t)NEXP_ * D_ * FF_;

  __bf16* wb_in  = (__bf16*)(ws + alloc(n_in_w  * 2));
  __bf16* wb_qkv = (__bf16*)(ws + alloc(n_qkv_w * 2));
  __bf16* wb_out = (__bf16*)(ws + alloc(n_out_w * 2));
  __bf16* wb_ff1 = (__bf16*)(ws + alloc(n_ff1_w * 2));
  __bf16* wb_ff2 = (__bf16*)(ws + alloc(n_ff2_w * 2));
  __bf16* wb_e1  = (__bf16*)(ws + alloc(n_e1_w  * 2));
  __bf16* wb_e2  = (__bf16*)(ws + alloc(n_e2_w  * 2));
  __bf16* wb_hc  = (__bf16*)(ws + alloc((size_t)64 * D_ * 2));
  __bf16* wb_hb  = (__bf16*)(ws + alloc((size_t)32 * D_ * 2));
  __bf16* xb     = (__bf16*)(ws + alloc((size_t)M_ * IN_ * 2));
  float*  h      = (float* )(ws + alloc((size_t)M_ * D_ * 4));
  __bf16* zb     = (__bf16*)(ws + alloc((size_t)M_ * D_ * 2));
  __bf16* big    = (__bf16*)(ws + alloc((size_t)M_ * FF_ * 2)); // qkv / ff-hidden / expert-hidden (aliased)
  __bf16* attnb  = (__bf16*)(ws + alloc((size_t)M_ * D_ * 2));
  __bf16* hb16   = (__bf16*)(ws + alloc((size_t)M_ * D_ * 2));
  float*  gated  = (float* )(ws + alloc((size_t)M_ * NEXP_ * 4));
  float*  moe    = (float* )(ws + alloc((size_t)M_ * D_ * 4));

  auto cvt = [&](const float* src, __bf16* dst, size_t n) {
    k_cvt_bf16<<<dim3((unsigned)((n + 255) / 256)), dim3(256), 0, stream>>>(src, dst, n);
  };
  auto gemm = [&](const __bf16* A, const __bf16* W, const float* bias,
                  float* Cf, __bf16* Cb, const float* resid,
                  const float* rscale, int rss, int M, int N, int K, int act) {
    dim3 g((N + 127) / 128, (M + 127) / 128);
    k_gemm_bf16<<<g, dim3(256), 0, stream>>>(A, W, bias, Cf, Cb, resid, rscale, rss, M, N, K, act);
  };

  // ---- weight + input conversion ----
  cvt(x,     xb,     (size_t)M_ * IN_);
  cvt(in_w,  wb_in,  n_in_w);
  cvt(qkv_w, wb_qkv, n_qkv_w);
  cvt(out_w, wb_out, n_out_w);
  cvt(ff1_w, wb_ff1, n_ff1_w);
  cvt(ff2_w, wb_ff2, n_ff2_w);
  cvt(e1_w,  wb_e1,  n_e1_w);
  cvt(e2_w,  wb_e2,  n_e2_w);
  cvt(hc_w,  wb_hc,  (size_t)64 * D_);
  cvt(hb_w,  wb_hb,  (size_t)32 * D_);

  // ---- input projection + positional encoding ----
  gemm(xb, wb_in, in_b, h, nullptr, nullptr, nullptr, 0, M_, D_, IN_, 0);
  k_add_pe<<<dim3((M_ * D_) / 256), dim3(256), 0, stream>>>(h);

  // ---- transformer layers ----
  for (int l = 0; l < L_; ++l) {
    k_layernorm_bf16<<<dim3(M_), dim3(256), 0, stream>>>(h, ln1_w + l * D_, ln1_b + l * D_, zb);
    gemm(zb, wb_qkv + (size_t)l * 3 * D_ * D_, qkv_b + l * 3 * D_,
         nullptr, big, nullptr, nullptr, 0, M_, 3 * D_, D_, 0);
    k_flash_attn<<<dim3(S_ / 16, H_, B_), dim3(32), 0, stream>>>(big, attnb);
    gemm(attnb, wb_out + (size_t)l * D_ * D_, out_b + l * D_,
         h, nullptr, h, nullptr, 0, M_, D_, D_, 0);                      // h += attn proj
    k_layernorm_bf16<<<dim3(M_), dim3(256), 0, stream>>>(h, ln2_w + l * D_, ln2_b + l * D_, zb);
    gemm(zb, wb_ff1 + (size_t)l * FF_ * D_, ff1_b + l * FF_,
         nullptr, big, nullptr, nullptr, 0, M_, FF_, D_, 1);             // GELU -> bf16
    gemm(big, wb_ff2 + (size_t)l * D_ * FF_, ff2_b + l * D_,
         h, nullptr, h, nullptr, 0, M_, D_, FF_, 0);                     // h += ffn
  }

  // ---- MoE ----
  cvt(h, hb16, (size_t)M_ * D_);
  k_gate<<<dim3(M_), dim3(32), 0, stream>>>(h, gate_w, gate_b, gated);
  hipMemsetAsync(moe, 0, (size_t)M_ * D_ * 4, stream);
  for (int n = 0; n < NEXP_; ++n) {
    gemm(hb16, wb_e1 + (size_t)n * FF_ * D_, e1_b + n * FF_,
         nullptr, big, nullptr, nullptr, 0, M_, FF_, D_, 2);             // ReLU -> bf16
    gemm(big, wb_e2 + (size_t)n * D_ * FF_, e2_b + n * D_,
         moe, nullptr, nullptr, gated + n, NEXP_, M_, D_, FF_, 0);       // moe += g_n * expert_n
  }

  // ---- final norm + heads (straight into d_out: [M,64] then [M,32]) ----
  k_layernorm_bf16<<<dim3(M_), dim3(256), 0, stream>>>(moe, fn_w, fn_b, zb);
  float* outf = (float*)d_out;
  gemm(zb, wb_hc, hc_b, outf,                    nullptr, nullptr, nullptr, 0, M_, 64, D_, 0);
  gemm(zb, wb_hb, hb_b, outf + (size_t)M_ * 64,  nullptr, nullptr, nullptr, 0, M_, 32, D_, 0);
}
